// DeepSetController_38817914422128
// MI455X (gfx1250) — matrix operand order repaired and checked by reading, not compiled, not run on hardware
//
#include <hip/hip_runtime.h>

typedef __attribute__((ext_vector_type(16))) _Float16 v16h;
typedef __attribute__((ext_vector_type(2)))  _Float16 h2;
typedef __attribute__((ext_vector_type(8)))  float    v8f;
typedef __attribute__((ext_vector_type(4)))  float    v4f;

#define NDATA        262144
#define XSTRIDE      136        // 8 self + 128 neighbor floats
#define WPB          8          // waves per block (256 threads, wave32)
#define BLOCKS       4096
#define TOTAL_WAVES  (WPB * BLOCKS)   // 32768
#define SPW          (NDATA / TOTAL_WAVES) // 8 samples per wave

static __device__ __forceinline__ v8f wmma_f16(v16h a, v16h b, v8f c) {
  // 8 args: (neg_a, A, neg_b, B, c_mod, C, reuse_a, reuse_b)
  return __builtin_amdgcn_wmma_f32_16x16x32_f16(false, a, false, b, (short)0, c,
                                                false, false);
}

static __device__ __forceinline__ float relu(float v) { return v > 0.f ? v : 0.f; }

__global__ __launch_bounds__(256) void deepset_kernel(
    const float* __restrict__ x,
    const float* __restrict__ phi_w1, const float* __restrict__ phi_b1,
    const float* __restrict__ phi_w2, const float* __restrict__ phi_b2,
    const float* __restrict__ rho_w1, const float* __restrict__ rho_b1,
    const float* __restrict__ rho_w2, const float* __restrict__ rho_b2,
    float* __restrict__ out)
{
  const int lane  = threadIdx.x & 31;
  const int wave  = threadIdx.x >> 5;
  const int gwave = blockIdx.x * WPB + wave;
  const int n     = lane & 15;   // column / N index inside a 16-wide tile
  const int g     = lane >> 4;   // lane half-group

  // H1 staged per wave: 16 rows x 32 f16, row-interleaved so (j, j+16) are
  // adjacent -> one 32B contiguous A-fragment load per lane.
  __shared__ __align__(64) _Float16 ldsH[WPB][16][32];
  __shared__ __align__(16) float    ldsS[WPB][16];

  // ---------------- one-time weight staging (per wave) ----------------
  // B for phi layer 1: physical K = d (0..7) only; group-1 lanes all zero.
  v16h b1t0 = {}; v16h b1t1 = {};
  if (g == 0) {
#pragma unroll
    for (int t = 0; t < 8; ++t) {
      b1t0[t] = (_Float16)phi_w1[n * 8 + t];          // W1[j][d],  j = n
      b1t1[t] = (_Float16)phi_w1[(n + 16) * 8 + t];   // W1[j][d],  j = n+16
    }
  }
  // B for phi layer 2 with the K-permutation that matches the interleaved
  // LDS layout of A (physical K p -> logical j):
  v16h b2m = {};
#pragma unroll
  for (int t = 0; t < 16; ++t) {
    int p = t + 16 * g;
    int j;
    j = (p >> 1) + 16 * (p & 1);
    b2m[t] = (_Float16)phi_w2[n * 32 + j];            // W2[o][j], o = n
  }

  const float b1a = phi_b1[n];
  const float b1b = phi_b1[n + 16];
  const float b2v = phi_b2[n];

  float rw[24];
  {
    const float* row = rho_w1 + lane * 24;            // rho_w1 is (32,24)
#pragma unroll
    for (int i = 0; i < 24; ++i) rw[i] = row[i];
  }
  const float rb1  = rho_b1[lane];
  const float w2a  = rho_w2[lane];                    // rho_w2 row 0
  const float w2b  = rho_w2[32 + lane];               // rho_w2 row 1
  const float rb20 = rho_b2[0];
  const float rb21 = rho_b2[1];

  const v8f zero = {};

  for (int it = 0; it < SPW; ++it) {
    const long sample = gwave + (long)it * TOTAL_WAVES;
    const float* xs   = x + sample * XSTRIDE;

    if (it + 1 < SPW) {
      const char* nxt = (const char*)(x + (sample + TOTAL_WAVES) * XSTRIDE);
      __builtin_prefetch(nxt + (lane & 7) * 64, 0, 0);  // global_prefetch_b8
    }

    // ---- phi layer 1: A = neighbor matrix 16x8 (K zero-padded to 32) ----
    v16h a1 = {};
    if (g == 0) {
      const float* nb = xs + 8;                 // nb[k][d] = x[8 + d*16 + k]
#pragma unroll
      for (int d = 0; d < 8; ++d)
        a1[d] = (_Float16)nb[d * 16 + n];       // lane n = neighbor row k
    }
    v8f d0 = wmma_f16(a1, b1t0, zero);          // H1[:, 0..15]  pre-act
    v8f d1 = wmma_f16(a1, b1t1, zero);          // H1[:,16..31]  pre-act

    // bias + relu + f16, interleaved store: (M, j=n) and (M, j=n+16) packed
    h2* hrow = (h2*)&ldsH[wave][0][0];
#pragma unroll
    for (int r = 0; r < 8; ++r) {
      int M = r + 8 * g;
      h2 hv;
      hv.x = (_Float16)relu(d0[r] + b1a);
      hv.y = (_Float16)relu(d1[r] + b1b);
      hrow[M * 16 + n] = hv;
    }
    __builtin_amdgcn_wave_barrier();
    asm volatile("s_wait_dscnt 0" ::: "memory");
    __builtin_amdgcn_wave_barrier();

    // ---- phi layer 2: A fragment = one contiguous 32B LDS read per lane ----
    v16h a2 = *(const v16h*)&ldsH[wave][n][g * 16];
    v8f  d2 = wmma_f16(a2, b2m, zero);          // H2[k][o], o = n

    // sum over the 16 neighbors (rows M)
    float acc = 0.f;
#pragma unroll
    for (int r = 0; r < 8; ++r) acc += relu(d2[r] + b2v);
    acc += __shfl_xor(acc, 16, 32);
    if (g == 0) ldsS[wave][n] = acc;
    __builtin_amdgcn_wave_barrier();
    asm volatile("s_wait_dscnt 0" ::: "memory");
    __builtin_amdgcn_wave_barrier();

    // ---- rho layer 1: lane l computes y[l] (24 -> 32) ----
    v4f s0 = *(const v4f*)(xs);
    v4f s1 = *(const v4f*)(xs + 4);
    const v4f* sm = (const v4f*)&ldsS[wave][0];
    v4f m0 = sm[0], m1 = sm[1], m2 = sm[2], m3 = sm[3];

    float y = rb1;
    y += s0.x * rw[0]  + s0.y * rw[1]  + s0.z * rw[2]  + s0.w * rw[3];
    y += s1.x * rw[4]  + s1.y * rw[5]  + s1.z * rw[6]  + s1.w * rw[7];
    y += m0.x * rw[8]  + m0.y * rw[9]  + m0.z * rw[10] + m0.w * rw[11];
    y += m1.x * rw[12] + m1.y * rw[13] + m1.z * rw[14] + m1.w * rw[15];
    y += m2.x * rw[16] + m2.y * rw[17] + m2.z * rw[18] + m2.w * rw[19];
    y += m3.x * rw[20] + m3.y * rw[21] + m3.z * rw[22] + m3.w * rw[23];
    y = relu(y);

    // ---- rho layer 2: (32 -> 2) via wave reduction ----
    float t0 = y * w2a;
    float t1 = y * w2b;
#pragma unroll
    for (int off = 16; off > 0; off >>= 1) {
      t0 += __shfl_xor(t0, off, 32);
      t1 += __shfl_xor(t1, off, 32);
    }
    if (lane == 0) {
      out[2 * sample]     = relu(t0 + rb20);
      out[2 * sample + 1] = relu(t1 + rb21);
    }
  }
}

extern "C" void kernel_launch(void* const* d_in, const int* in_sizes, int n_in,
                              void* d_out, int out_size, void* d_ws, size_t ws_size,
                              hipStream_t stream) {
  (void)in_sizes; (void)n_in; (void)out_size; (void)d_ws; (void)ws_size;
  const float* x      = (const float*)d_in[0];
  const float* phi_w1 = (const float*)d_in[1];
  const float* phi_b1 = (const float*)d_in[2];
  const float* phi_w2 = (const float*)d_in[3];
  const float* phi_b2 = (const float*)d_in[4];
  const float* rho_w1 = (const float*)d_in[5];
  const float* rho_b1 = (const float*)d_in[6];
  const float* rho_w2 = (const float*)d_in[7];
  const float* rho_b2 = (const float*)d_in[8];
  float* out = (float*)d_out;

  deepset_kernel<<<dim3(BLOCKS), dim3(256), 0, stream>>>(
      x, phi_w1, phi_b1, phi_w2, phi_b2, rho_w1, rho_b1, rho_w2, rho_b2, out);
}
